// FourDNet_25194278158993
// MI455X (gfx1250) — matrix operand order repaired and checked by reading, not compiled, not run on hardware
//
#include <hip/hip_runtime.h>
#include <hip/hip_bf16.h>

typedef __attribute__((ext_vector_type(16))) _Float16 v16h;
typedef __attribute__((ext_vector_type(8)))  float    v8f;

// ---------------------------------------------------------------------------
// Generic f16-WMMA GEMM:  C[M,N] = act( A[M,K]@B[K,N] + bias[N] + rowbias )
// A is f32 (converted to f16 on load).  B is f32 row-major (K x N).
// A row addressing supports a batched layout:
//   addr(m) = a_base + (m/rpb)*batch_stride + (m%rpb)*lda
// rowbias (optional): x += rowbias[(m/bias_rpb)*N + n]
// act: 0 = none, 1 = sigmoid
// One wave32 computes a 16x16 tile via v_wmma_f32_16x16x32_f16.
// Block = 128 threads = 4 waves = 32x32 C tile.
// n_clean (N%32==0): B tile staged cooperatively into LDS with
// global_load_async_to_lds_b128 (ASYNCcnt) and read back per-fragment.
// Out-of-range loads use clamped indices (branchless); only stores guarded.
// ---------------------------------------------------------------------------
struct GemmParams {
  const float* A;
  const float* B;
  const float* bias;     // may be null
  const float* rowbias;  // may be null
  float*       C;
  int M, N, K;
  int lda;
  int rpb;
  long long batch_stride;
  long long a_base;
  int ldc;
  int bias_rpb;
  int act;
  int n_clean;
};

__global__ void gemm_wmma_f16(GemmParams p) {
  __shared__ float sB[32 * 32];   // one 32(K) x 32(N) f32 B tile
  const int tid  = threadIdx.x;
  const int warp = tid >> 5;
  const int lane = tid & 31;
  const int half = lane >> 4;      // K-half selector for A/B fragments
  const int lr   = lane & 15;

  const int mBase = blockIdx.y * 32 + (warp >> 1) * 16;
  const int nTile = blockIdx.x * 32;
  const int nBase = nTile + (warp & 1) * 16;

  v8f acc = {};

  // ---- A addressing (clamped, branchless) ----
  const int m  = mBase + lr;
  const int me = (m < p.M) ? m : (p.M - 1);
  const float* arow = p.A + p.a_base
                    + (long long)(me / p.rpb) * p.batch_stride
                    + (long long)(me % p.rpb) * p.lda;

  // ---- B addressing ----
  const int n  = nBase + lr;
  const int ne = (n < p.N) ? n : (p.N - 1);   // fallback path clamp
  // async staging: 128 threads stage full 32x32 tile, 2 x b128 each
  const int sr  = tid >> 3;                   // 0..15   (K row)
  const int sc4 = (tid & 7) * 4;              // 0,4,...,28 (N col group)
  const unsigned lds0 = (unsigned)(size_t)(&sB[sr * 32 + sc4]);
  const unsigned lds1 = (unsigned)(size_t)(&sB[(sr + 16) * 32 + sc4]);
  const int nloc = (warp & 1) * 16 + lr;      // fragment column within tile

  for (int k0 = 0; k0 < p.K; k0 += 32) {
    // A fragment: lanes<16 take K {0..15}, lanes>=16 {16..31}
    v16h af;
    const int kh = k0 + half * 16;
#pragma unroll
    for (int i = 0; i < 8; ++i) af[i]     = (_Float16)arow[kh + i];
#pragma unroll
    for (int i = 0; i < 8; ++i) af[8 + i] = (_Float16)arow[kh + 8 + i];
    if (k0 + 32 < p.K) __builtin_prefetch(arow + kh + 32, 0, 1);

    // B fragment: lane holds column n; lanes<16 K {0..15}, lanes>=16 {16..31}
    v16h bf;
    if (p.n_clean) {
      const float* g0 = p.B + (long long)(k0 + sr) * p.N + nTile + sc4;
      const float* g1 = g0 + 16LL * p.N;
      asm volatile("global_load_async_to_lds_b128 %0, %1, off"
                   :: "v"(lds0), "v"((unsigned long long)(size_t)g0)
                   : "memory");
      asm volatile("global_load_async_to_lds_b128 %0, %1, off"
                   :: "v"(lds1), "v"((unsigned long long)(size_t)g1)
                   : "memory");
      asm volatile("s_wait_asynccnt 0" ::: "memory");
      __syncthreads();
#pragma unroll
      for (int i = 0; i < 16; ++i)
        bf[i] = (_Float16)sB[(half * 16 + i) * 32 + nloc];
      __syncthreads();   // reads done before next iteration's async writes
    } else {
      const int kb = k0 + half * 16;
#pragma unroll
      for (int i = 0; i < 16; ++i)
        bf[i] = (_Float16)p.B[(long long)(kb + i) * p.N + ne];
    }

    acc = __builtin_amdgcn_wmma_f32_16x16x32_f16(false, af, false, bf,
                                                 (short)0, acc, false, false);
  }

  // C/D layout: lane<16 -> rows mBase+0..7, lane>=16 -> rows mBase+8..15
  const int col = nBase + lr;
  if (col >= p.N) return;
  const float bv = p.bias ? p.bias[col] : 0.0f;
  const int rb = mBase + half * 8;
#pragma unroll
  for (int r = 0; r < 8; ++r) {
    const int row = rb + r;
    if (row < p.M) {
      float x = acc[r] + bv;
      if (p.rowbias) x += p.rowbias[(long long)(row / p.bias_rpb) * p.N + col];
      if (p.act == 1) x = 1.0f / (1.0f + __expf(-x));
      p.C[(long long)row * p.ldc + col] = x;
    }
  }
}

// ---------------------------------------------------------------------------
// conv1: 1 -> 64 channels, 3x3, stride 2, pad 1.  K=9 -> scalar VALU kernel.
// In: d (B,1,256,128) f32.  Out: (B,64,128,64) f16.
// ---------------------------------------------------------------------------
__global__ void conv1_kernel(const float* __restrict__ in,
                             const float* __restrict__ w,   // (64,1,3,3)
                             const float* __restrict__ bias,
                             _Float16* __restrict__ out,
                             int B, int Hin, int Win, int Cout, int Hout, int Wout) {
  long long o = (long long)blockIdx.x * blockDim.x + threadIdx.x;
  long long total = (long long)B * Cout * Hout * Wout;
  if (o >= total) return;
  int wo = (int)(o % Wout);
  int ho = (int)((o / Wout) % Hout);
  int co = (int)((o / ((long long)Wout * Hout)) % Cout);
  int b  = (int)(o / ((long long)Wout * Hout * Cout));
  float acc = bias[co];
  const float* wc = w + co * 9;
  const float* ib = in + (long long)b * Hin * Win;
#pragma unroll
  for (int rr = 0; rr < 3; ++rr) {
#pragma unroll
    for (int ss = 0; ss < 3; ++ss) {
      int hi = ho * 2 - 1 + rr;
      int wi = wo * 2 - 1 + ss;
      if ((unsigned)hi < (unsigned)Hin && (unsigned)wi < (unsigned)Win)
        acc += wc[rr * 3 + ss] * ib[(long long)hi * Win + wi];
    }
  }
  out[o] = (_Float16)acc;
}

// ---------------------------------------------------------------------------
// Implicit-GEMM WMMA conv: 3x3, stride 2, pad 1, f16 in, f32 weights.
// M-dim = 16 consecutive output-W positions, N-dim = 16 out channels,
// K = Cin*9 (576 for conv2, 1152 for conv3; both multiples of 32).
// mode 0: write NCHW f16.   mode 1: write (B, Hout*Wout, Cout) f32 (DEPT).
// Block = 128 threads = 4 waves along N.
// ---------------------------------------------------------------------------
__device__ inline _Float16 conv_fetch(const _Float16* __restrict__ in,
                                      long long base_b, int Cin, int Hin, int Win,
                                      int ho, int wo, int kk) {
  int cin = kk / 9;
  int rs  = kk - cin * 9;
  int rr  = rs / 3;
  int ss  = rs - rr * 3;
  int hi  = ho * 2 - 1 + rr;
  int wi  = wo * 2 - 1 + ss;
  if ((unsigned)hi >= (unsigned)Hin || (unsigned)wi >= (unsigned)Win)
    return (_Float16)0.0f;
  return in[base_b + ((long long)cin * Hin + hi) * Win + wi];
}

__global__ void conv_wmma_kernel(const _Float16* __restrict__ in,
                                 const float* __restrict__ wts,  // (Cout, Cin*9)
                                 const float* __restrict__ bias,
                                 _Float16* __restrict__ out_nchw,
                                 float* __restrict__ out_dept,
                                 int B, int Cin, int Hin, int Win,
                                 int Cout, int Hout, int Wout, int mode) {
  const int warp = threadIdx.x >> 5;
  const int lane = threadIdx.x & 31;
  const int half = lane >> 4;
  const int lr   = lane & 15;

  const int K  = Cin * 9;
  const int WT = Wout >> 4;
  const int mt = blockIdx.y;
  const int wt = mt % WT;
  const int t  = mt / WT;
  const int ho = t % Hout;
  const int b  = t / Hout;
  const int nBase = blockIdx.x * 64 + warp * 16;

  const long long base_b = (long long)b * Cin * Hin * Win;
  const int wo_a = wt * 16 + lr;            // this lane's A-row output position
  const int n    = nBase + lr;              // this lane's B column (out channel)
  const float* bcol = wts + (long long)n * K;

  v8f acc = {};
  for (int k0 = 0; k0 < K; k0 += 32) {
    v16h af;
    v16h bf;
    const int kh = k0 + half * 16;
#pragma unroll
    for (int i = 0; i < 8; ++i)
      af[i] = conv_fetch(in, base_b, Cin, Hin, Win, ho, wo_a, kh + i);
#pragma unroll
    for (int i = 0; i < 8; ++i)
      af[8 + i] = conv_fetch(in, base_b, Cin, Hin, Win, ho, wo_a, kh + 8 + i);
    const int kb = k0 + half * 16;
#pragma unroll
    for (int i = 0; i < 16; ++i) bf[i] = (_Float16)bcol[kb + i];
    acc = __builtin_amdgcn_wmma_f32_16x16x32_f16(false, af, false, bf,
                                                 (short)0, acc, false, false);
  }

  const int ch = nBase + lr;                // output channel for this lane
  const float bv = bias[ch];
  const int rb = wt * 16 + half * 8;        // output-W base for this lane's rows
#pragma unroll
  for (int r = 0; r < 8; ++r) {
    const int wo = rb + r;
    const float x = acc[r] + bv;
    if (mode == 0) {
      out_nchw[(((long long)b * Cout + ch) * Hout + ho) * Wout + wo] = (_Float16)x;
    } else {
      out_dept[((long long)b * (Hout * Wout) + ho * Wout + wo) * Cout + ch] = x;
    }
  }
}

// ---------------------------------------------------------------------------
// MGC[n] = b_merge[n] + sum_j g_depth[j] * W_merge[j,n]   (top half of merge)
// ---------------------------------------------------------------------------
__global__ void mgc_kernel(const float* __restrict__ g_depth,
                           const float* __restrict__ W_merge,  // (256,128)
                           const float* __restrict__ b_merge,
                           float* __restrict__ mgc) {
  int n = threadIdx.x;   // 128 threads
  float s = b_merge[n];
  for (int j = 0; j < 128; ++j) s += g_depth[j] * W_merge[j * 128 + n];
  mgc[n] = s;
}

// ---------------------------------------------------------------------------
// Attention / bilinear-gather pooling.  One block per (b,n); 128 threads.
// SEL: (B*128, 72) sigmoid already applied.  V: (B, 512, 128).
// r2d[b,n,c] = (1/24) * sum_k attn_k * coeff_k *
//              (xv0*yv0*f0 + xv0*yv1*f1 + xv1*yv0*f2 + xv1*yv1*f3)
// ---------------------------------------------------------------------------
__global__ void interp_kernel(const float* __restrict__ SEL,
                              const float* __restrict__ V,
                              float* __restrict__ r2d) {
  __shared__ float s_sel[72];
  __shared__ float s_attn[24];
  __shared__ int   s_idx[24][4];
  __shared__ float s_wgt[24][4];

  const int row = blockIdx.x;           // b*128 + n
  const int c   = threadIdx.x;          // channel

  if (c < 72) s_sel[c] = SEL[(long long)row * 72 + c];
  __syncthreads();

  if (c == 0) {
    float mx = -1e30f;
    for (int k = 0; k < 24; ++k) mx = fmaxf(mx, s_sel[48 + k]);
    float ssum = 0.0f;
    for (int k = 0; k < 24; ++k) { float e = __expf(s_sel[48 + k] - mx); s_attn[k] = e; ssum += e; }
    float inv = 1.0f / ssum;
    for (int k = 0; k < 24; ++k) s_attn[k] *= inv;
  }
  if (c < 24) {
    const int k = c;
    const float lx = s_sel[2 * k];
    const float ly = s_sel[2 * k + 1];
    const float x1f = floorf(lx * 15.0f);
    const float x2f = fminf(x1f + 1.0f, 15.0f);
    const float y1f = floorf(ly * 31.0f);
    const float y2f = fminf(y1f + 1.0f, 31.0f);
    const int xi1 = (int)x1f, xi2 = (int)x2f, yi1 = (int)y1f, yi2 = (int)y2f;
    s_idx[k][0] = xi1 * 16 + yi2;
    s_idx[k][1] = xi2 * 16 + yi2;
    s_idx[k][2] = xi2 * 16 + yi1;
    s_idx[k][3] = xi1 * 16 + yi1;
    const float x = lx * 16.0f, y = ly * 32.0f;
    const float w_ = x2f - x1f, h_ = y2f - y1f;
    const bool  deg = (h_ == 0.0f) || (w_ == 0.0f);
    const float denom = (h_ * w_ == 0.0f) ? 1.0f : h_ * w_;
    const float coeff = deg ? 2.0f : 1.0f / denom;
    const float xv0 = (w_ == 0.0f) ? 1.0f : (x2f - x);
    const float xv1 = (w_ == 0.0f) ? 1.0f : (x - x1f);
    const float yv0 = (h_ == 0.0f) ? 1.0f : (y2f - y);
    const float yv1 = (h_ == 0.0f) ? 1.0f : (y - y1f);
    s_wgt[k][0] = coeff * xv0 * yv0;
    s_wgt[k][1] = coeff * xv0 * yv1;
    s_wgt[k][2] = coeff * xv1 * yv0;
    s_wgt[k][3] = coeff * xv1 * yv1;
  }
  __syncthreads();

  const float* vb = V + (long long)(row >> 7) * 512 * 128;
  float acc = 0.0f;
#pragma unroll 4
  for (int k = 0; k < 24; ++k) {
    acc += s_attn[k] * (s_wgt[k][0] * vb[s_idx[k][0] * 128 + c] +
                        s_wgt[k][1] * vb[s_idx[k][1] * 128 + c] +
                        s_wgt[k][2] * vb[s_idx[k][2] * 128 + c] +
                        s_wgt[k][3] * vb[s_idx[k][3] * 128 + c]);
  }
  r2d[(long long)row * 128 + c] = acc * (1.0f / 24.0f);
}

// ---------------------------------------------------------------------------
// FE[b,c] = mean_n r2d[b,n,c]; also writes FE into d_out tail.
// ---------------------------------------------------------------------------
__global__ void reduce_fe_kernel(const float* __restrict__ r2d,
                                 float* __restrict__ fe,
                                 float* __restrict__ out_tail) {
  const int b = blockIdx.x;
  const int c = threadIdx.x;   // 128
  float s = 0.0f;
  const float* rb = r2d + (long long)b * 128 * 128;
  for (int n = 0; n < 128; ++n) s += rb[(long long)n * 128 + c];
  const float v = s * (1.0f / 128.0f);
  fe[b * 128 + c] = v;
  out_tail[b * 128 + c] = v;
}

// ---------------------------------------------------------------------------
// Host-side launcher
// ---------------------------------------------------------------------------
static void launch_gemm(hipStream_t s, const float* A, const float* B,
                        const float* bias, const float* rowbias, float* C,
                        int M, int N, int K, int lda, int rpb,
                        long long bstride, long long abase,
                        int bias_rpb, int act) {
  GemmParams p;
  p.A = A; p.B = B; p.bias = bias; p.rowbias = rowbias; p.C = C;
  p.M = M; p.N = N; p.K = K; p.lda = lda; p.rpb = rpb;
  p.batch_stride = bstride; p.a_base = abase; p.ldc = N;
  p.bias_rpb = bias_rpb ? bias_rpb : 1; p.act = act;
  p.n_clean = (N % 32 == 0) ? 1 : 0;
  dim3 grid((N + 31) / 32, (M + 31) / 32);
  gemm_wmma_f16<<<grid, 128, 0, s>>>(p);
}

static inline char* carve(char* ws, size_t& off, size_t bytes) {
  char* p = ws + off;
  off += (bytes + 255) & ~(size_t)255;
  return p;
}

extern "C" void kernel_launch(void* const* d_in, const int* in_sizes, int n_in,
                              void* d_out, int out_size, void* d_ws, size_t ws_size,
                              hipStream_t stream) {
  (void)in_sizes; (void)n_in; (void)out_size; (void)ws_size;
  const float* features = (const float*)d_in[0];   // (64,129,768)
  const float* dmap     = (const float*)d_in[1];   // (64,1,256,128)
  const float* W_rg     = (const float*)d_in[2];   // (768,128)
  const float* b_rg     = (const float*)d_in[3];
  const float* W_rl     = (const float*)d_in[4];   // (768,128)
  const float* b_rl     = (const float*)d_in[5];
  const float* conv1_w  = (const float*)d_in[6];   // (64,1,3,3)
  const float* conv1_b  = (const float*)d_in[7];
  const float* conv2_w  = (const float*)d_in[8];   // (128,64,3,3)
  const float* conv2_b  = (const float*)d_in[9];
  const float* conv3_w  = (const float*)d_in[10];  // (128,128,3,3)
  const float* conv3_b  = (const float*)d_in[11];
  const float* g_depth  = (const float*)d_in[12];  // (128)
  const float* W_merge  = (const float*)d_in[13];  // (256,128)
  const float* b_merge  = (const float*)d_in[14];
  const float* W_q      = (const float*)d_in[15];  // (256,128)
  const float* b_q      = (const float*)d_in[16];
  const float* W_v      = (const float*)d_in[17];  // (128,128)
  const float* b_v      = (const float*)d_in[18];
  const float* W_sel    = (const float*)d_in[19];  // (128,72)
  const float* b_sel    = (const float*)d_in[20];
  const float* W_cls    = (const float*)d_in[21];  // (128,751)
  float* out = (float*)d_out;                      // cls (64*751) then fe (64*128)

  const int B = 64;
  const long long FB = 129LL * 768;   // feature batch stride

  size_t off = 0;
  char* ws = (char*)d_ws;
  float*    GF   = (float*)   carve(ws, off, sizeof(float) * 64 * 128);
  float*    LF   = (float*)   carve(ws, off, sizeof(float) * 8192 * 128);
  float*    QG   = (float*)   carve(ws, off, sizeof(float) * 64 * 128);
  float*    Q    = (float*)   carve(ws, off, sizeof(float) * 8192 * 128);
  float*    SEL  = (float*)   carve(ws, off, sizeof(float) * 8192 * 72);
  _Float16* C1   = (_Float16*)carve(ws, off, sizeof(_Float16) * (size_t)64 * 64 * 128 * 64);
  _Float16* C2   = (_Float16*)carve(ws, off, sizeof(_Float16) * (size_t)64 * 128 * 64 * 32);
  float*    DEPT = (float*)   carve(ws, off, sizeof(float) * (size_t)64 * 512 * 128);
  float*    MGC  = (float*)   carve(ws, off, sizeof(float) * 128);
  float*    MG   = (float*)   carve(ws, off, sizeof(float) * (size_t)64 * 512 * 128);
  float*    V    = (float*)   carve(ws, off, sizeof(float) * (size_t)64 * 512 * 128);
  float*    R2D  = (float*)   carve(ws, off, sizeof(float) * 8192 * 128);
  float*    FE   = (float*)   carve(ws, off, sizeof(float) * 64 * 128);

  const int BIG = 0x40000000;

  // 1) GF = features[:,0] @ W_rg + b_rg            (64 x 128, K=768)
  launch_gemm(stream, features, W_rg, b_rg, nullptr, GF,
              64, 128, 768, /*lda*/768, /*rpb*/1, /*bstride*/FB, /*abase*/0,
              1, 0);
  // 2) LF = features[:,1:] @ W_rl + b_rl           (8192 x 128, K=768)
  launch_gemm(stream, features, W_rl, b_rl, nullptr, LF,
              8192, 128, 768, 768, /*rpb*/128, FB, /*abase*/768,
              1, 0);
  // 3) QG = GF @ W_q[0:128] + b_q                  (64 x 128, K=128)
  launch_gemm(stream, GF, W_q, b_q, nullptr, QG,
              64, 128, 128, 128, BIG, 0, 0, 1, 0);
  // 4) Q = LF @ W_q[128:256] + rowbias(QG)         (8192 x 128, K=128)
  launch_gemm(stream, LF, W_q + 128 * 128, nullptr, QG, Q,
              8192, 128, 128, 128, BIG, 0, 0, /*bias_rpb*/128, 0);
  // 5) SEL = sigmoid(Q @ W_sel + b_sel)            (8192 x 72, K=128)
  launch_gemm(stream, Q, W_sel, b_sel, nullptr, SEL,
              8192, 72, 128, 128, BIG, 0, 0, 1, /*act*/1);

  // 6) conv1 (scalar): d (64,1,256,128) -> C1 (64,64,128,64) f16
  {
    long long total = (long long)64 * 64 * 128 * 64;
    int blocks = (int)((total + 255) / 256);
    conv1_kernel<<<blocks, 256, 0, stream>>>(dmap, conv1_w, conv1_b, C1,
                                             B, 256, 128, 64, 128, 64);
  }
  // 7) conv2 (implicit-GEMM WMMA): C1 -> C2 (64,128,64,32) f16
  {
    dim3 grid(128 / 64, B * 64 * (32 / 16));   // (2, 8192)
    conv_wmma_kernel<<<grid, 128, 0, stream>>>(C1, conv2_w, conv2_b, C2, nullptr,
                                               B, 64, 128, 64, 128, 64, 32, /*mode*/0);
  }
  // 8) conv3 (implicit-GEMM WMMA): C2 -> DEPT (64,512,128) f32 (transposed write)
  {
    dim3 grid(128 / 64, B * 32 * (16 / 16));   // (2, 2048)
    conv_wmma_kernel<<<grid, 128, 0, stream>>>(C2, conv3_w, conv3_b, nullptr, DEPT,
                                               B, 128, 64, 32, 128, 32, 16, /*mode*/1);
  }

  // 9) MGC = g_depth @ W_merge[0:128] + b_merge    (128,)
  mgc_kernel<<<1, 128, 0, stream>>>(g_depth, W_merge, b_merge, MGC);
  // 10) MG = DEPT @ W_merge[128:256] + MGC         (32768 x 128, K=128)
  launch_gemm(stream, DEPT, W_merge + 128 * 128, MGC, nullptr, MG,
              32768, 128, 128, 128, BIG, 0, 0, 1, 0);
  // 11) V = MG @ W_v + b_v                          (32768 x 128, K=128)
  launch_gemm(stream, MG, W_v, b_v, nullptr, V,
              32768, 128, 128, 128, BIG, 0, 0, 1, 0);

  // 12) attention / bilinear gather -> R2D (8192 x 128)
  interp_kernel<<<8192, 128, 0, stream>>>(SEL, V, R2D);
  // 13) FE = mean_n R2D; also write d_out tail
  reduce_fe_kernel<<<64, 128, 0, stream>>>(R2D, FE, out + 64 * 751);
  // 14) cls = FE @ W_cls -> d_out head              (64 x 751, K=128)
  launch_gemm(stream, FE, W_cls, nullptr, nullptr, out,
              64, 751, 128, 128, BIG, 0, 0, 1, 0);
}
